// MultiHeadAttention_91027536871977
// MI455X (gfx1250) — matrix operand order repaired and checked by reading, not compiled, not run on hardware
//
#include <hip/hip_runtime.h>

// ---------------------------------------------------------------------------
// Cosine-similarity multi-head attention for MI455X (gfx1250).
// - All GEMM stages on v_wmma_f32_16x16x32_bf16 (f32 accumulate).
// - Fragment loads are 2x ds_load_b128 (fragment-friendly LDS orientation).
// - K-loop GEMMs (proj / out-proj) stage tiles with the Tensor Data Mover
//   (tensor_load_to_lds + s_wait_tensorcnt), double-buffered in LDS.
// B=8, S=1024, D=1024, H=16, F=64.
// ---------------------------------------------------------------------------

typedef __attribute__((ext_vector_type(16))) __bf16 v16bf;
typedef __attribute__((ext_vector_type(8)))  __bf16 v8bf;
typedef __attribute__((ext_vector_type(8)))  float  v8f;
typedef __attribute__((ext_vector_type(4)))  unsigned tdm_g0_t;
typedef __attribute__((ext_vector_type(8)))  int      tdm_g1_t;
typedef __attribute__((ext_vector_type(4)))  int      tdm_g2_t;

static constexpr int B_ = 8, S_ = 1024, D_ = 1024, H_ = 16, F_ = 64;

__device__ inline __bf16 f2bf(float f) {
  unsigned u = __builtin_bit_cast(unsigned, f);
  unsigned r = u + 0x7FFFu + ((u >> 16) & 1u);   // round-to-nearest-even
  unsigned short h = (unsigned short)(r >> 16);
  return __builtin_bit_cast(__bf16, h);
}

__device__ inline v8f wmma_bf16(v16bf a, v16bf b, v8f c) {
  return __builtin_amdgcn_wmma_f32_16x16x32_bf16(false, a, false, b,
                                                 (short)0, c, false, false);
}

__device__ inline v16bf cat8(v8bf lo, v8bf hi) {
  return __builtin_shufflevector(lo, hi, 0, 1, 2, 3, 4, 5, 6, 7,
                                 8, 9, 10, 11, 12, 13, 14, 15);
}

// A fragment (16x32 bf16, wave32): lane<16 -> M=lane, K in {0..7,16..23};
// lane>=16 -> M=lane-16, K in {8..15,24..31}. Source row-major [m][k].
__device__ inline v16bf load_frag_a(const __bf16* p, int ld, int m0, int k0) {
  const int lane = threadIdx.x & 31;
  const __bf16* row = p + (size_t)(m0 + (lane & 15)) * ld + k0 + ((lane >> 4) << 3);
  return cat8(*(const v8bf*)row, *(const v8bf*)(row + 16));
}

// B fragment (32x16): B[k][n] = T[n][k] with T row-major [n][k], stride ld.
__device__ inline v16bf load_frag_bt(const __bf16* p, int ld, int k0, int n0) {
  const int lane = threadIdx.x & 31;
  const __bf16* row = p + (size_t)(n0 + (lane & 15)) * ld + k0 + ((lane >> 4) << 3);
  return cat8(*(const v8bf*)row, *(const v8bf*)(row + 16));
}

// ---------------------------------------------------------------------------
// Tensor Data Mover: 2D tile (tile_w x tile_h bf16 elements, row stride
// `stride` elements) from global -> LDS, packed row-major (stride tile_w).
// D# per CDNA5 ISA ch.8: group0 {count=1, lds_addr, global_addr, type=2},
// group1 {data_size=1(2B), tensor_dim0/1, tile_dim0/1, tensor_dim0_stride}.
// 6-arg builtin form (clang-23 / therock): trailing int32x8 group zero-filled.
// Issued wave-uniformly (TDM ignores EXEC); tracked by TENSORcnt.
// ---------------------------------------------------------------------------
__device__ inline void tdm_load_2d(const void* lds_dst, const void* gsrc,
                                   unsigned tile_w, unsigned tile_h,
                                   unsigned stride) {
  const unsigned long long ga = (unsigned long long)gsrc;
  const unsigned lds = (unsigned)(unsigned long long)lds_dst;  // LDS aperture low bits
  const unsigned td0 = stride;          // elements per row (OOB bound)
  const unsigned td1 = tile_h;

  tdm_g0_t g0;
  g0[0] = 1u;                                          // count=1, user D#
  g0[1] = lds;                                         // lds_addr (bytes)
  g0[2] = (unsigned)(ga & 0xFFFFFFFFu);                // global_addr[31:0]
  g0[3] = (unsigned)((ga >> 32) & 0x1FFFFFFu) | (2u << 30);  // [56:32] | type=2

  tdm_g1_t g1;
  g1[0] = (int)(1u << 16);                             // data_size=1 (2 bytes)
  g1[1] = (int)((td0 & 0xFFFFu) << 16);                // tensor_dim0 lo
  g1[2] = (int)((td0 >> 16) | ((td1 & 0xFFFFu) << 16));// tensor_dim0 hi | dim1 lo
  g1[3] = (int)((td1 >> 16) | (tile_w << 16));         // dim1 hi | tile_dim0
  g1[4] = (int)(tile_h & 0xFFFFu);                     // tile_dim1
  g1[5] = (int)stride;                                 // tensor_dim0_stride lo32
  g1[6] = 0;                                           // stride hi | dim1_stride lo
  g1[7] = 0;

  tdm_g2_t gz = {};                                    // 2D: groups 2/3 unused
  tdm_g1_t gz8 = {};                                   // trailing group (6-arg form)
  __builtin_amdgcn_tensor_load_to_lds(g0, g1, gz, gz, gz8, 0);
}

// ---------------------------------------------------------------------------
// Prep kernels: bf16 conversions (+ transposed copies).
// ---------------------------------------------------------------------------
__global__ __launch_bounds__(256) void k_cvt(const float* __restrict__ src,
                                             __bf16* __restrict__ dst, int n4) {
  const int i = blockIdx.x * 256 + threadIdx.x;
  if (i >= n4) return;
  const float4 v = ((const float4*)src)[i];
  union { __bf16 h[4]; uint2 u; } o;
  o.h[0] = f2bf(v.x); o.h[1] = f2bf(v.y); o.h[2] = f2bf(v.z); o.h[3] = f2bf(v.w);
  ((uint2*)dst)[i] = o.u;
}

__global__ __launch_bounds__(256) void k_cvt_t(const float* __restrict__ src,
                                               __bf16* __restrict__ dst,
                                               int rows, int cols) {
  const size_t base = (size_t)blockIdx.y * rows * cols;
  const int i = blockIdx.x * 256 + threadIdx.x;
  if (i >= rows * cols) return;
  const int r = i / cols, c = i % cols;
  dst[base + (size_t)c * rows + r] = f2bf(src[base + i]);
}

__global__ __launch_bounds__(256) void k_xtr(const __bf16* __restrict__ src,
                                             __bf16* __restrict__ dst) {
  const size_t base = (size_t)blockIdx.y * S_ * F_;
  const int i = blockIdx.x * 256 + threadIdx.x;   // over S*F = 65536
  const int s = i >> 6, f = i & 63;
  dst[base + (size_t)f * S_ + s] = src[base + i];
}

// ---------------------------------------------------------------------------
// Kernel 1: X[b,h] = sin[b] @ Wx[h] + bx[h]  (bf16 out) + 1/||x|| per row.
// TDM double-buffered staging; grid (S/128, B*H), 8 waves.
// ---------------------------------------------------------------------------
__global__ __launch_bounds__(256) void k_proj(const __bf16* __restrict__ sinbf,
                                              const __bf16* __restrict__ WxT,
                                              const float* __restrict__ bx,
                                              __bf16* __restrict__ Xbf,
                                              float* __restrict__ rnorm) {
  __shared__ __align__(16) __bf16 lA[2][128 * 32];
  __shared__ __align__(16) __bf16 lBt[2][64 * 32];

  const int bh = blockIdx.y, b = bh / H_, h = bh % H_;
  const int row0 = blockIdx.x * 128;
  const int tid = threadIdx.x, wave = tid >> 5, lane = tid & 31;
  const int nin = lane & 15, mhalf = (lane >> 4) * 8;

  const __bf16* Ab = sinbf + ((size_t)b * S_ + row0) * D_;
  const __bf16* Bb = WxT + (size_t)h * F_ * D_;

  if (wave == 0) {
    tdm_load_2d(lA[0], Ab, 32, 128, D_);
    tdm_load_2d(lBt[0], Bb, 32, 64, D_);
  }

  v8f acc[4] = {};
  constexpr int NT = D_ / 32;
  for (int kc = 0; kc < NT; ++kc) {
    const int cur = kc & 1;
    if (wave == 0) {
      if (kc + 1 < NT) {
        tdm_load_2d(lA[cur ^ 1], Ab + (kc + 1) * 32, 32, 128, D_);
        tdm_load_2d(lBt[cur ^ 1], Bb + (kc + 1) * 32, 32, 64, D_);
        __builtin_amdgcn_s_wait_tensorcnt(2);   // oldest two (cur) done
      } else {
        __builtin_amdgcn_s_wait_tensorcnt(0);
      }
    }
    __syncthreads();
    const v16bf a  = load_frag_a(lA[cur], 32, wave * 16, 0);
    const v16bf b0 = load_frag_bt(lBt[cur], 32, 0, 0);
    const v16bf b1 = load_frag_bt(lBt[cur], 32, 0, 16);
    const v16bf b2 = load_frag_bt(lBt[cur], 32, 0, 32);
    const v16bf b3 = load_frag_bt(lBt[cur], 32, 0, 48);
    acc[0] = wmma_bf16(a, b0, acc[0]);
    acc[1] = wmma_bf16(a, b1, acc[1]);
    acc[2] = wmma_bf16(a, b2, acc[2]);
    acc[3] = wmma_bf16(a, b3, acc[3]);
    __syncthreads();
  }

  // epilogue: bias, squared-norm reduction, bf16 store
  float ss[8] = {};
#pragma unroll
  for (int n = 0; n < 4; ++n) {
    const float bias = bx[h * F_ + n * 16 + nin];
#pragma unroll
    for (int r = 0; r < 8; ++r) {
      const float v = acc[n][r] + bias;
      acc[n][r] = v;
      ss[r] += v * v;
    }
  }
  const size_t base = (size_t)bh * S_ + row0 + wave * 16;
#pragma unroll
  for (int r = 0; r < 8; ++r) {
    float s = ss[r];
    s += __shfl_xor(s, 1, 32);
    s += __shfl_xor(s, 2, 32);
    s += __shfl_xor(s, 4, 32);
    s += __shfl_xor(s, 8, 32);
    if (nin == 0) rnorm[base + mhalf + r] = rsqrtf(s > 0.f ? s : 1.f);
  }
#pragma unroll
  for (int n = 0; n < 4; ++n)
#pragma unroll
    for (int r = 0; r < 8; ++r)
      Xbf[(base + mhalf + r) * F_ + n * 16 + nin] = f2bf(acc[n][r]);
}

// ---------------------------------------------------------------------------
// Kernel 2: online-softmax stats (row max m, sumexp l) of cosine scores.
// ---------------------------------------------------------------------------
__global__ __launch_bounds__(256) void k_stats(const __bf16* __restrict__ Xbf,
                                               const float* __restrict__ rnorm,
                                               float* __restrict__ Ms,
                                               float* __restrict__ Ls) {
  __shared__ __align__(16) __bf16 lXs[64 * 64];
  __shared__ __align__(16) __bf16 lXt[64 * 64];
  __shared__ float lrnt[64];
  __shared__ float redM[8][16];
  __shared__ float redL[8][16];

  const int bh = blockIdx.y;
  const int s0 = blockIdx.x * 64;
  const int tid = threadIdx.x, wave = tid >> 5, lane = tid & 31;
  const int rowtile = wave & 3, grp = wave >> 2;
  const int nin = lane & 15, mhalf = (lane >> 4) * 8;
  const float NEG_INF = -__builtin_inff();

  const __bf16* Xh = Xbf + (size_t)bh * S_ * F_;
  const float* rn = rnorm + (size_t)bh * S_;

  {
    const int4* src = (const int4*)(Xh + (size_t)s0 * F_);
    int4* dst = (int4*)lXs;
    dst[tid] = src[tid];
    dst[tid + 256] = src[tid + 256];
  }

  float rn_s[8];
#pragma unroll
  for (int r = 0; r < 8; ++r) rn_s[r] = rn[s0 + rowtile * 16 + mhalf + r];

  float runM[8], runL[8];
#pragma unroll
  for (int r = 0; r < 8; ++r) { runM[r] = NEG_INF; runL[r] = 0.f; }

  for (int t0 = 0; t0 < S_; t0 += 64) {
    __syncthreads();
    {
      const int4* src = (const int4*)(Xh + (size_t)t0 * F_);
      int4* dst = (int4*)lXt;
      dst[tid] = src[tid];
      dst[tid + 256] = src[tid + 256];
      if (tid < 64) lrnt[tid] = rn[t0 + tid];
    }
    __syncthreads();

    const v16bf a0 = load_frag_a(lXs, 64, rowtile * 16, 0);
    const v16bf a1 = load_frag_a(lXs, 64, rowtile * 16, 32);
    const int c0 = grp * 2, c1 = grp * 2 + 1;
    const v16bf bA0 = load_frag_bt(lXt, 64, 0, c0 * 16);
    const v16bf bA1 = load_frag_bt(lXt, 64, 32, c0 * 16);
    const v16bf bB0 = load_frag_bt(lXt, 64, 0, c1 * 16);
    const v16bf bB1 = load_frag_bt(lXt, 64, 32, c1 * 16);
    v8f d0 = {}, d1 = {};
    d0 = wmma_bf16(a0, bA0, d0);
    d0 = wmma_bf16(a1, bA1, d0);
    d1 = wmma_bf16(a0, bB0, d1);
    d1 = wmma_bf16(a1, bB1, d1);

#pragma unroll
    for (int j = 0; j < 2; ++j) {
      const v8f d = j ? d1 : d0;
      const int c = j ? c1 : c0;
      const float rnc = lrnt[c * 16 + nin];
#pragma unroll
      for (int r = 0; r < 8; ++r) {
        float v = d[r] * rn_s[r] * rnc;
        if (v == 0.f) v = NEG_INF;                 // reference quirk
        float vm = v;
        vm = fmaxf(vm, __shfl_xor(vm, 1, 32));
        vm = fmaxf(vm, __shfl_xor(vm, 2, 32));
        vm = fmaxf(vm, __shfl_xor(vm, 4, 32));
        vm = fmaxf(vm, __shfl_xor(vm, 8, 32));
        const float newM = fmaxf(runM[r], vm);
        float p = (newM == NEG_INF) ? 0.f : __expf(v - newM);
        p += __shfl_xor(p, 1, 32);
        p += __shfl_xor(p, 2, 32);
        p += __shfl_xor(p, 4, 32);
        p += __shfl_xor(p, 8, 32);
        const float scale = (runM[r] == NEG_INF) ? 0.f : __expf(runM[r] - newM);
        runL[r] = runL[r] * scale + p;
        runM[r] = newM;
      }
    }
  }

  __syncthreads();
  if (nin == 0) {
#pragma unroll
    for (int r = 0; r < 8; ++r) {
      redM[wave][mhalf + r] = runM[r];
      redL[wave][mhalf + r] = runL[r];
    }
  }
  __syncthreads();
  if (tid < 64) {
    const int w1 = tid >> 4, ri = tid & 15;
    const float m1 = redM[w1][ri], l1 = redL[w1][ri];
    const float m2 = redM[w1 + 4][ri], l2 = redL[w1 + 4][ri];
    const float m = fmaxf(m1, m2);
    float l = 0.f;
    if (m != NEG_INF) l = l1 * __expf(m1 - m) + l2 * __expf(m2 - m);
    Ms[(size_t)bh * S_ + s0 + tid] = m;
    Ls[(size_t)bh * S_ + s0 + tid] = l;
  }
}

// ---------------------------------------------------------------------------
// Kernel 3: O = softmax(scores) @ X, heads concatenated (bf16).
// ---------------------------------------------------------------------------
__global__ __launch_bounds__(256) void k_av(const __bf16* __restrict__ Xbf,
                                            const __bf16* __restrict__ XbfT,
                                            const float* __restrict__ rnorm,
                                            const float* __restrict__ Ms,
                                            const float* __restrict__ Ls,
                                            __bf16* __restrict__ Obf) {
  __shared__ __align__(16) __bf16 lXs[64 * 64];
  __shared__ __align__(16) __bf16 lXt[64 * 64];   // [t][f] for score B-frags
  __shared__ __align__(16) __bf16 lXtT[64 * 64];  // [f][t] for AV B-frags
  __shared__ __align__(16) __bf16 lP[64 * 64];
  __shared__ float lrnt[64];

  const int bh = blockIdx.y, b = bh >> 4, h = bh & 15;
  const int s0 = blockIdx.x * 64;
  const int tid = threadIdx.x, wave = tid >> 5, lane = tid & 31;
  const int rowtile = wave & 3, grp = wave >> 2;
  const int nin = lane & 15, mhalf = (lane >> 4) * 8;
  const float NEG_INF = -__builtin_inff();

  const __bf16* Xh = Xbf + (size_t)bh * S_ * F_;
  const __bf16* XhT = XbfT + (size_t)bh * F_ * S_;
  const float* rn = rnorm + (size_t)bh * S_;

  {
    const int4* src = (const int4*)(Xh + (size_t)s0 * F_);
    int4* dst = (int4*)lXs;
    dst[tid] = src[tid];
    dst[tid + 256] = src[tid + 256];
  }

  float rn_s[8], mrow[8], linv[8];
#pragma unroll
  for (int r = 0; r < 8; ++r) {
    const int row = s0 + rowtile * 16 + mhalf + r;
    rn_s[r] = rn[row];
    mrow[r] = Ms[(size_t)bh * S_ + row];
    const float l = Ls[(size_t)bh * S_ + row];
    linv[r] = (l > 0.f) ? 1.f / l : 0.f;
  }

  v8f o0 = {}, o1 = {};

  for (int t0 = 0; t0 < S_; t0 += 64) {
    __syncthreads();
    {
      const int4* src = (const int4*)(Xh + (size_t)t0 * F_);
      int4* dst = (int4*)lXt;
      dst[tid] = src[tid];
      dst[tid + 256] = src[tid + 256];
      // transposed tile: row f -> XhT[f][t0..t0+63], 8 int4 per row
      int4* dstT = (int4*)lXtT;
      {
        const int f = tid >> 3, q = tid & 7;
        dstT[tid] = *(const int4*)(XhT + (size_t)f * S_ + t0 + q * 8);
        const int f2 = (tid + 256) >> 3;
        dstT[tid + 256] = *(const int4*)(XhT + (size_t)f2 * S_ + t0 + q * 8);
      }
      if (tid < 64) lrnt[tid] = rn[t0 + tid];
    }
    __syncthreads();

    const v16bf a0 = load_frag_a(lXs, 64, rowtile * 16, 0);
    const v16bf a1 = load_frag_a(lXs, 64, rowtile * 16, 32);
    const int c0 = grp * 2, c1 = grp * 2 + 1;
    const v16bf bA0 = load_frag_bt(lXt, 64, 0, c0 * 16);
    const v16bf bA1 = load_frag_bt(lXt, 64, 32, c0 * 16);
    const v16bf bB0 = load_frag_bt(lXt, 64, 0, c1 * 16);
    const v16bf bB1 = load_frag_bt(lXt, 64, 32, c1 * 16);
    v8f d0 = {}, d1 = {};
    d0 = wmma_bf16(a0, bA0, d0);
    d0 = wmma_bf16(a1, bA1, d0);
    d1 = wmma_bf16(a0, bB0, d1);
    d1 = wmma_bf16(a1, bB1, d1);

#pragma unroll
    for (int j = 0; j < 2; ++j) {
      const v8f d = j ? d1 : d0;
      const int c = j ? c1 : c0;
      const float rnc = lrnt[c * 16 + nin];
#pragma unroll
      for (int r = 0; r < 8; ++r) {
        float v = d[r] * rn_s[r] * rnc;
        if (v == 0.f) v = NEG_INF;                 // reference quirk
        const float p = (mrow[r] == NEG_INF) ? 0.f
                         : __expf(v - mrow[r]) * linv[r];
        lP[(rowtile * 16 + mhalf + r) * 64 + c * 16 + nin] = f2bf(p);
      }
    }
    __syncthreads();

    const v16bf pa0 = load_frag_a(lP, 64, rowtile * 16, 0);
    const v16bf pa1 = load_frag_a(lP, 64, rowtile * 16, 32);
    const int f0 = (grp * 2) * 16, f1 = (grp * 2 + 1) * 16;
    const v16bf v00 = load_frag_bt(lXtT, 64, 0, f0);
    const v16bf v01 = load_frag_bt(lXtT, 64, 32, f0);
    const v16bf v10 = load_frag_bt(lXtT, 64, 0, f1);
    const v16bf v11 = load_frag_bt(lXtT, 64, 32, f1);
    o0 = wmma_bf16(pa0, v00, o0);
    o0 = wmma_bf16(pa1, v01, o0);
    o1 = wmma_bf16(pa0, v10, o1);
    o1 = wmma_bf16(pa1, v11, o1);
  }

#pragma unroll
  for (int r = 0; r < 8; ++r) {
    const int row = s0 + rowtile * 16 + mhalf + r;
    const size_t base = ((size_t)b * S_ + row) * (H_ * F_) + h * F_;
    Obf[base + (grp * 2) * 16 + nin] = f2bf(o0[r]);
    Obf[base + (grp * 2 + 1) * 16 + nin] = f2bf(o1[r]);
  }
}

// ---------------------------------------------------------------------------
// Kernel 4: out = O @ Wp + bp (f32). TDM double-buffered staging.
// ---------------------------------------------------------------------------
__global__ __launch_bounds__(256) void k_out(const __bf16* __restrict__ Obf,
                                             const __bf16* __restrict__ WpT,
                                             const float* __restrict__ bp,
                                             float* __restrict__ out) {
  __shared__ __align__(16) __bf16 lA[2][128 * 32];
  __shared__ __align__(16) __bf16 lBt[2][64 * 32];
  const int row0 = blockIdx.x * 128;
  const int col0 = blockIdx.y * 64;
  const int tid = threadIdx.x, wave = tid >> 5, lane = tid & 31;
  const int nin = lane & 15, mhalf = (lane >> 4) * 8;
  constexpr int K = H_ * F_;   // 1024
  constexpr int N = D_;        // 1024

  const __bf16* Ab = Obf + (size_t)row0 * K;
  const __bf16* Bb = WpT + (size_t)col0 * K;

  if (wave == 0) {
    tdm_load_2d(lA[0], Ab, 32, 128, K);
    tdm_load_2d(lBt[0], Bb, 32, 64, K);
  }

  v8f acc[4] = {};
  constexpr int NT = K / 32;
  for (int kc = 0; kc < NT; ++kc) {
    const int cur = kc & 1;
    if (wave == 0) {
      if (kc + 1 < NT) {
        tdm_load_2d(lA[cur ^ 1], Ab + (kc + 1) * 32, 32, 128, K);
        tdm_load_2d(lBt[cur ^ 1], Bb + (kc + 1) * 32, 32, 64, K);
        __builtin_amdgcn_s_wait_tensorcnt(2);
      } else {
        __builtin_amdgcn_s_wait_tensorcnt(0);
      }
    }
    __syncthreads();
    const v16bf a  = load_frag_a(lA[cur], 32, wave * 16, 0);
    const v16bf b0 = load_frag_bt(lBt[cur], 32, 0, 0);
    const v16bf b1 = load_frag_bt(lBt[cur], 32, 0, 16);
    const v16bf b2 = load_frag_bt(lBt[cur], 32, 0, 32);
    const v16bf b3 = load_frag_bt(lBt[cur], 32, 0, 48);
    acc[0] = wmma_bf16(a, b0, acc[0]);
    acc[1] = wmma_bf16(a, b1, acc[1]);
    acc[2] = wmma_bf16(a, b2, acc[2]);
    acc[3] = wmma_bf16(a, b3, acc[3]);
    __syncthreads();
  }

#pragma unroll
  for (int n = 0; n < 4; ++n) {
    const int col = col0 + n * 16 + nin;
    const float bias = bp[col];
#pragma unroll
    for (int r = 0; r < 8; ++r) {
      const int row = row0 + wave * 16 + mhalf + r;
      out[(size_t)row * N + col] = acc[n][r] + bias;
    }
  }
}

// ---------------------------------------------------------------------------
extern "C" void kernel_launch(void* const* d_in, const int* in_sizes, int n_in,
                              void* d_out, int out_size, void* d_ws, size_t ws_size,
                              hipStream_t stream) {
  (void)in_sizes; (void)n_in; (void)out_size; (void)ws_size;
  const float* sin_ = (const float*)d_in[0];
  // d_in[1] = mask (all-true; only its presence matters) — unused
  const float* Wx = (const float*)d_in[2];
  const float* bx = (const float*)d_in[3];
  const float* Wp = (const float*)d_in[4];
  const float* bp = (const float*)d_in[5];
  float* out = (float*)d_out;

  char* w = (char*)d_ws;
  constexpr size_t MB = 1u << 20;
  __bf16* sinbf = (__bf16*)(w);                   // 16 MiB  [b][s][d]
  __bf16* Xbf   = (__bf16*)(w + 16 * MB);         // 16 MiB  [bh][s][f]
  __bf16* XbfT  = (__bf16*)(w + 32 * MB);         // 16 MiB  [bh][f][s]
  __bf16* Obf   = (__bf16*)(w + 48 * MB);         // 16 MiB  [b][s][h*F]
  __bf16* WxT   = (__bf16*)(w + 64 * MB);         //  2 MiB  [h][f][d]
  __bf16* WpT   = (__bf16*)(w + 66 * MB);         //  2 MiB  [n][k]
  float* rnorm  = (float*)(w + 68 * MB);          // 0.5 MiB
  float* Ms     = (float*)(w + 68 * MB + (1u << 19));
  float* Ls     = (float*)(w + 68 * MB + (2u << 19));

  const dim3 blk(256);
  k_cvt  <<<dim3((B_ * S_ * D_ / 4 + 255) / 256), blk, 0, stream>>>(sin_, sinbf, B_ * S_ * D_ / 4);
  k_cvt_t<<<dim3((D_ * F_ + 255) / 256, H_), blk, 0, stream>>>(Wx, WxT, D_, F_);
  k_cvt_t<<<dim3((D_ * D_ + 255) / 256, 1), blk, 0, stream>>>(Wp, WpT, H_ * F_, D_);

  k_proj <<<dim3(S_ / 128, B_ * H_), blk, 0, stream>>>(sinbf, WxT, bx, Xbf, rnorm);
  k_xtr  <<<dim3(S_ * F_ / 256, B_ * H_), blk, 0, stream>>>(Xbf, XbfT);
  k_stats<<<dim3(S_ / 64, B_ * H_), blk, 0, stream>>>(Xbf, rnorm, Ms, Ls);
  k_av   <<<dim3(S_ / 64, B_ * H_), blk, 0, stream>>>(Xbf, XbfT, rnorm, Ms, Ls, Obf);
  k_out  <<<dim3(B_ * S_ / 128, D_ / 64), blk, 0, stream>>>(Obf, WpT, bp, out);
}
